// CSPAttention_46093589021345
// MI455X (gfx1250) — matrix operand order repaired and checked by reading, not compiled, not run on hardware
//
#include <hip/hip_runtime.h>

typedef __bf16 bf16_t;
typedef __attribute__((ext_vector_type(16))) __bf16 v16bf;
typedef __attribute__((ext_vector_type(8)))  __bf16 v8bf;
typedef __attribute__((ext_vector_type(8)))  float  v8f;

#define M_ROWS 8192      // B*L
#define D_MODEL 1024
#define D_ATTN 512
#define N_HEADS 8
#define HEAD_DIM 64
#define SEQ_L 2048

union AFrag { v16bf v; v8bf h[2]; };

// ---- WMMA helper: D = A(16x32 bf16) * B(32x16 bf16) + C(16x16 f32) ----
__device__ inline v8f wmma_bf16(v16bf a, v16bf b, v8f c) {
  return __builtin_amdgcn_wmma_f32_16x16x32_bf16(false, a, false, b, (short)0, c, false, false);
}

// A fragment (16x32, bf16), row-major source. Contiguous k order (matches B):
// lanes 0-15: row=lane, K={0..15}
// lanes16-31: row=lane-16, K={16..31}
__device__ inline v16bf load_frag_a(const bf16_t* __restrict__ base, int ld,
                                    int row0, int k0, int lane) {
  const bf16_t* p = base + (size_t)(row0 + (lane & 15)) * ld + k0;
  int koff = (lane >> 4) << 4;  // 0 or 16
  AFrag f;
  f.h[0] = *(const v8bf*)(p + koff);
  f.h[1] = *(const v8bf*)(p + koff + 8);
  return f.v;
}

// B fragment (32x16, bf16). Element(k,n). Lane holds one column
// n = lane&15; lanes 0-15 hold K=0..15, lanes 16-31 hold K=16..31 (contiguous).
// Source is row-major with "rows" indexed by n (W is (N,K): column of W^T ==
// row of W, contiguous in memory -> one 32-byte load per lane).
__device__ inline v16bf load_frag_b(const bf16_t* __restrict__ base, int ld,
                                    int n0, int k0, int lane) {
  const bf16_t* p = base + (size_t)(n0 + (lane & 15)) * ld + k0 + ((lane >> 4) << 4);
  AFrag f;
  f.h[0] = *(const v8bf*)(p);
  f.h[1] = *(const v8bf*)(p + 8);
  return f.v;
}

// ---- CDNA5 async copy: 16 bytes per lane, global -> LDS (ASYNCcnt) --------
__device__ inline void async_copy16(const bf16_t* gsrc, bf16_t* ldst) {
  unsigned loff =
      (unsigned)(size_t)(__attribute__((address_space(3))) bf16_t*)ldst;
  asm volatile("global_load_async_to_lds_b128 %0, %1, off"
               :: "v"(loff), "v"(gsrc) : "memory");
}

// ---------------- generic GEMM: out = X(MxK) * W^T(KxN) + bias ------------
// X row-major bf16 (ld=K), W row-major (N,K) bf16, out row-major (ld=out_ld).
// Block 256 thr = 8 waves arranged 2(M)x4(N); wave tile 32x32 -> block 64x128.
// K is consumed in 64-wide slabs double-buffered through LDS via
// global_load_async_to_lds_b128; next slab's fetch overlaps current WMMAs.
template <typename OutT>
__global__ __launch_bounds__(256) void gemm_wmma(
    const bf16_t* __restrict__ X, const bf16_t* __restrict__ W,
    const float* __restrict__ bias, OutT* __restrict__ out,
    int K, int out_ld) {
  __shared__ __align__(16) bf16_t As[2][64][72];
  __shared__ __align__(16) bf16_t Bs[2][128][72];

  const int lane = threadIdx.x & 31;
  const int wid  = threadIdx.x >> 5;
  const int wm = wid >> 2, wn = wid & 3;
  const int rowg = blockIdx.y * 64;
  const int colg = blockIdx.x * 128;

  v8f acc[2][2];
  acc[0][0] = {}; acc[0][1] = {}; acc[1][0] = {}; acc[1][1] = {};

  auto issue = [&](int slab, int buf) {
    const int k0 = slab * 64;
#pragma unroll
    for (int j = 0; j < 2; ++j) {           // A tile: 64 rows x 64 k = 512 x 16B
      int ch = threadIdx.x + j * 256;
      int r = ch >> 3, kc = (ch & 7) * 8;
      async_copy16(X + (size_t)(rowg + r) * K + k0 + kc, &As[buf][r][kc]);
    }
#pragma unroll
    for (int j = 0; j < 4; ++j) {           // B tile: 128 rows x 64 k = 1024 x 16B
      int ch = threadIdx.x + j * 256;
      int r = ch >> 3, kc = (ch & 7) * 8;
      async_copy16(W + (size_t)(colg + r) * K + k0 + kc, &Bs[buf][r][kc]);
    }
  };

  const int nslab = K >> 6;
  issue(0, 0);
  for (int i = 0; i < nslab; ++i) {
    __syncthreads();                         // prior slab's readers done
    if (i + 1 < nslab) {
      issue(i + 1, (i + 1) & 1);             // prefetch next slab (6 async ops)
      asm volatile("s_wait_asynccnt 0x6" ::: "memory");  // slab i complete
    } else {
      asm volatile("s_wait_asynccnt 0x0" ::: "memory");
    }
    __syncthreads();                         // slab i visible to all waves
    const bf16_t* Ab = &As[i & 1][0][0];
    const bf16_t* Bb = &Bs[i & 1][0][0];
#pragma unroll
    for (int kk = 0; kk < 64; kk += 32) {
      v16bf a0 = load_frag_a(Ab, 72, wm * 32,      kk, lane);
      v16bf a1 = load_frag_a(Ab, 72, wm * 32 + 16, kk, lane);
      v16bf b0 = load_frag_b(Bb, 72, wn * 32,      kk, lane);
      v16bf b1 = load_frag_b(Bb, 72, wn * 32 + 16, kk, lane);
      acc[0][0] = wmma_bf16(a0, b0, acc[0][0]);
      acc[0][1] = wmma_bf16(a0, b1, acc[0][1]);
      acc[1][0] = wmma_bf16(a1, b0, acc[1][0]);
      acc[1][1] = wmma_bf16(a1, b1, acc[1][1]);
    }
  }

  const int row0 = rowg + wm * 32;
  const int colb = colg + wn * 32 + (lane & 15);
  const int rofs = (lane >> 4) << 3;
#pragma unroll
  for (int nj = 0; nj < 2; ++nj) {
    float bv = bias[colb + nj * 16];
#pragma unroll
    for (int mi = 0; mi < 2; ++mi)
#pragma unroll
      for (int i = 0; i < 8; ++i) {
        int r = row0 + mi * 16 + i + rofs;
        out[(size_t)r * out_ld + colb + nj * 16] = (OutT)(acc[mi][nj][i] + bv);
      }
  }
}

// ---------------- flash attention (per b,h; 64 q-rows per block) ----------
// Q,K,V: (8192 x 512) bf16, head h at cols [64h,64h+64). O same layout.
__global__ __launch_bounds__(128) void attn_wmma(
    const bf16_t* __restrict__ Q, const bf16_t* __restrict__ K,
    const bf16_t* __restrict__ V, bf16_t* __restrict__ O) {
  __shared__ __align__(16) bf16_t Vt[64][72];        // V^T tile: [d][key]
  __shared__ __align__(16) bf16_t Plds[4][16][72];   // per-wave P staging

  const int lane = threadIdx.x & 31;
  const int wid  = threadIdx.x >> 5;                 // 0..3
  const int h    = blockIdx.y;
  const int b    = blockIdx.z;
  const int rowbase = b * SEQ_L;
  const int qrow0   = blockIdx.x * 64 + wid * 16;    // local in L
  const int hoff    = h * HEAD_DIM;
  const int rofs    = (lane >> 4) << 3;

  v16bf qa[2];
  qa[0] = load_frag_a(Q, D_ATTN, rowbase + qrow0, hoff, lane);
  qa[1] = load_frag_a(Q, D_ATTN, rowbase + qrow0, hoff + 32, lane);

  float mrow[8], lrow[8];
  v8f o[4];
#pragma unroll
  for (int i = 0; i < 8; ++i) { mrow[i] = -1e30f; lrow[i] = 0.f; }
  o[0] = {}; o[1] = {}; o[2] = {}; o[3] = {};

  for (int kb = 0; kb < SEQ_L / 64; ++kb) {
    const int key0 = kb * 64;
    __syncthreads();  // protect Vt from previous iteration's readers
    // cooperative transpose-load of V tile (64 keys x 64 d) into Vt[d][key]
    for (int t = threadIdx.x; t < 64 * 8; t += 128) {
      int key = t >> 3;
      int dch = (t & 7) * 8;
      v8bf vv = *(const v8bf*)(V + (size_t)(rowbase + key0 + key) * D_ATTN + hoff + dch);
#pragma unroll
      for (int e = 0; e < 8; ++e) Vt[dch + e][key] = vv[e];
    }
    __syncthreads();

    // S = Q K^T / 8 : 4 tiles of 16 keys
    v8f st[4];
#pragma unroll
    for (int nt = 0; nt < 4; ++nt) {
      v8f s = {};
      s = wmma_bf16(qa[0], load_frag_b(K, D_ATTN, rowbase + key0 + nt * 16, hoff, lane), s);
      s = wmma_bf16(qa[1], load_frag_b(K, D_ATTN, rowbase + key0 + nt * 16, hoff + 32, lane), s);
      st[nt] = s * 0.125f;
    }

    // online softmax: row max over 64 keys (16 lanes per half-wave)
    float mnew[8];
#pragma unroll
    for (int i = 0; i < 8; ++i) {
      float m = mrow[i];
#pragma unroll
      for (int nt = 0; nt < 4; ++nt) m = fmaxf(m, st[nt][i]);
      mnew[i] = m;
    }
#pragma unroll
    for (int off = 8; off >= 1; off >>= 1)
#pragma unroll
      for (int i = 0; i < 8; ++i) mnew[i] = fmaxf(mnew[i], __shfl_xor(mnew[i], off, 32));

    float rsum[8];
#pragma unroll
    for (int i = 0; i < 8; ++i) {
      float alpha = __expf(mrow[i] - mnew[i]);
      mrow[i] = mnew[i];
      lrow[i] *= alpha;
      rsum[i] = 0.f;
#pragma unroll
      for (int dt = 0; dt < 4; ++dt) o[dt][i] = o[dt][i] * alpha;
    }
#pragma unroll
    for (int nt = 0; nt < 4; ++nt)
#pragma unroll
      for (int i = 0; i < 8; ++i) {
        float p = __expf(st[nt][i] - mrow[i]);
        st[nt][i] = p;
        rsum[i] += p;
      }
#pragma unroll
    for (int off = 8; off >= 1; off >>= 1)
#pragma unroll
      for (int i = 0; i < 8; ++i) rsum[i] += __shfl_xor(rsum[i], off, 32);
#pragma unroll
    for (int i = 0; i < 8; ++i) lrow[i] += rsum[i];

    // stage P (C layout) -> LDS -> reload as A fragments (wave-private)
#pragma unroll
    for (int nt = 0; nt < 4; ++nt)
#pragma unroll
      for (int i = 0; i < 8; ++i)
        Plds[wid][i + rofs][nt * 16 + (lane & 15)] = (bf16_t)st[nt][i];

    // O += P * V  (A from Plds, B from Vt: column n=d, contiguous keys)
#pragma unroll
    for (int kc = 0; kc < 2; ++kc) {
      const bf16_t* pp = &Plds[wid][lane & 15][kc * 32];
      AFrag pa;
      pa.h[0] = *(const v8bf*)(pp + ((lane >> 4) << 4));
      pa.h[1] = *(const v8bf*)(pp + ((lane >> 4) << 4) + 8);
#pragma unroll
      for (int dt = 0; dt < 4; ++dt) {
        const bf16_t* vp = &Vt[dt * 16 + (lane & 15)][kc * 32 + ((lane >> 4) << 4)];
        AFrag bf_;
        bf_.h[0] = *(const v8bf*)vp;
        bf_.h[1] = *(const v8bf*)(vp + 8);
        o[dt] = wmma_bf16(pa.v, bf_.v, o[dt]);
      }
    }
  }

#pragma unroll
  for (int dt = 0; dt < 4; ++dt)
#pragma unroll
    for (int i = 0; i < 8; ++i) {
      int r = rowbase + qrow0 + i + rofs;
      O[(size_t)r * D_ATTN + hoff + dt * 16 + (lane & 15)] = (bf16_t)(o[dt][i] / lrow[i]);
    }
}

// ---------------- depthwise conv-3 on channels [512,1024) ------------------
__global__ void conv_dw(const float* __restrict__ q, const float* __restrict__ w,
                        const float* __restrict__ bias, bf16_t* __restrict__ out) {
  int idx = blockIdx.x * blockDim.x + threadIdx.x;
  if (idx >= M_ROWS * 512) return;
  int c = idx & 511;
  int blr = idx >> 9;
  int l = blr & (SEQ_L - 1);
  const float* xp = q + (size_t)blr * D_MODEL + D_ATTN + c;
  float acc = bias[c] + w[c * 3 + 1] * xp[0];
  if (l > 0)          acc += w[c * 3 + 0] * xp[-D_MODEL];
  if (l < SEQ_L - 1)  acc += w[c * 3 + 2] * xp[D_MODEL];
  out[idx] = (bf16_t)acc;
}

// ---------------- f32 -> bf16 (with optional column slice) -----------------
__global__ void cast_slice(const float* __restrict__ src, bf16_t* __restrict__ dst,
                           int rows, int cols, int src_ld, int src_off) {
  int idx = blockIdx.x * blockDim.x + threadIdx.x;
  if (idx >= rows * cols) return;
  int r = idx / cols, c = idx - r * cols;
  dst[idx] = (bf16_t)src[(size_t)r * src_ld + src_off + c];
}

// ---------------- residual + LayerNorm -------------------------------------
__global__ __launch_bounds__(256) void add_ln(
    const float* __restrict__ q, const float* __restrict__ y,
    const float* __restrict__ gamma, const float* __restrict__ beta,
    float* __restrict__ out) {
  __shared__ float xs[D_MODEL];
  __shared__ float red[16];
  const int row = blockIdx.x;
  const float* qp = q + (size_t)row * D_MODEL;
  const float* yp = y + (size_t)row * D_MODEL;
  float s = 0.f, ss = 0.f;
  for (int i = threadIdx.x; i < D_MODEL; i += 256) {
    float v = qp[i] + yp[i];
    xs[i] = v; s += v; ss += v * v;
  }
#pragma unroll
  for (int off = 16; off >= 1; off >>= 1) {
    s += __shfl_xor(s, off, 32);
    ss += __shfl_xor(ss, off, 32);
  }
  int w = threadIdx.x >> 5, lane = threadIdx.x & 31;
  if (lane == 0) { red[w] = s; red[8 + w] = ss; }
  __syncthreads();
  if (threadIdx.x == 0) {
    float ts = 0.f, tss = 0.f;
    for (int i = 0; i < 8; ++i) { ts += red[i]; tss += red[8 + i]; }
    red[0] = ts; red[8] = tss;
  }
  __syncthreads();
  float mu  = red[0] * (1.f / D_MODEL);
  float var = red[8] * (1.f / D_MODEL) - mu * mu;
  float rs  = rsqrtf(var + 1e-5f);
  for (int i = threadIdx.x; i < D_MODEL; i += 256)
    out[(size_t)row * D_MODEL + i] = (xs[i] - mu) * rs * gamma[i] + beta[i];
}

extern "C" void kernel_launch(void* const* d_in, const int* in_sizes, int n_in,
                              void* d_out, int out_size, void* d_ws, size_t ws_size,
                              hipStream_t stream) {
  (void)in_sizes; (void)n_in; (void)out_size; (void)ws_size;
  const float* queries = (const float*)d_in[0];
  const float* keys    = (const float*)d_in[1];
  const float* values  = (const float*)d_in[2];
  const float* Wq = (const float*)d_in[3];  const float* bq = (const float*)d_in[4];
  const float* Wk = (const float*)d_in[5];  const float* bk = (const float*)d_in[6];
  const float* Wv = (const float*)d_in[7];  const float* bv = (const float*)d_in[8];
  const float* Wo_attn = (const float*)d_in[9];  const float* bo_attn = (const float*)d_in[10];
  const float* conv_w = (const float*)d_in[11];  const float* conv_b = (const float*)d_in[12];
  const float* Wo_conv = (const float*)d_in[13]; const float* bo_conv = (const float*)d_in[14];
  const float* Wf = (const float*)d_in[15]; const float* bfin = (const float*)d_in[16];
  const float* gamma = (const float*)d_in[17]; const float* beta = (const float*)d_in[18];

  char* ws = (char*)d_ws;
  size_t off = 0;
  auto alloc = [&](size_t bytes) -> char* {
    char* p = ws + off;
    off += (bytes + 255) & ~(size_t)255;
    return p;
  };
  const size_t half = (size_t)M_ROWS * D_ATTN;  // 8192*512
  bf16_t* Xq  = (bf16_t*)alloc(half * 2);
  bf16_t* Xk  = (bf16_t*)alloc(half * 2);
  bf16_t* Xv  = (bf16_t*)alloc(half * 2);
  bf16_t* Wqb = (bf16_t*)alloc(512 * 512 * 2);
  bf16_t* Wkb = (bf16_t*)alloc(512 * 512 * 2);
  bf16_t* Wvb = (bf16_t*)alloc(512 * 512 * 2);
  bf16_t* Woab= (bf16_t*)alloc(512 * 512 * 2);
  bf16_t* Wocb= (bf16_t*)alloc(512 * 512 * 2);
  bf16_t* Wfb = (bf16_t*)alloc(1024 * 1024 * 2);
  bf16_t* Qb  = (bf16_t*)alloc(half * 2);
  bf16_t* Kb  = (bf16_t*)alloc(half * 2);
  bf16_t* Vb  = (bf16_t*)alloc(half * 2);
  bf16_t* Ao  = (bf16_t*)alloc(half * 2);
  bf16_t* Co  = (bf16_t*)alloc(half * 2);
  bf16_t* Cat = (bf16_t*)alloc((size_t)M_ROWS * D_MODEL * 2);
  float*  Y   = (float*)alloc((size_t)M_ROWS * D_MODEL * 4);

  const int CB = 256;
  // 1. bf16 casts
  cast_slice<<<(M_ROWS * 512 + CB - 1) / CB, CB, 0, stream>>>(queries, Xq, M_ROWS, 512, 1024, 0);
  cast_slice<<<(M_ROWS * 512 + CB - 1) / CB, CB, 0, stream>>>(keys,    Xk, M_ROWS, 512, 1024, 0);
  cast_slice<<<(M_ROWS * 512 + CB - 1) / CB, CB, 0, stream>>>(values,  Xv, M_ROWS, 512, 1024, 0);
  cast_slice<<<(512 * 512 + CB - 1) / CB, CB, 0, stream>>>(Wq, Wqb, 512, 512, 512, 0);
  cast_slice<<<(512 * 512 + CB - 1) / CB, CB, 0, stream>>>(Wk, Wkb, 512, 512, 512, 0);
  cast_slice<<<(512 * 512 + CB - 1) / CB, CB, 0, stream>>>(Wv, Wvb, 512, 512, 512, 0);
  cast_slice<<<(512 * 512 + CB - 1) / CB, CB, 0, stream>>>(Wo_attn, Woab, 512, 512, 512, 0);
  cast_slice<<<(512 * 512 + CB - 1) / CB, CB, 0, stream>>>(Wo_conv, Wocb, 512, 512, 512, 0);
  cast_slice<<<(1024 * 1024 + CB - 1) / CB, CB, 0, stream>>>(Wf, Wfb, 1024, 1024, 1024, 0);

  // 2. Q/K/V projections (8192x512x512)
  dim3 g512(512 / 128, M_ROWS / 64);
  gemm_wmma<bf16_t><<<g512, 256, 0, stream>>>(Xq, Wqb, bq, Qb, 512, 512);
  gemm_wmma<bf16_t><<<g512, 256, 0, stream>>>(Xk, Wkb, bk, Kb, 512, 512);
  gemm_wmma<bf16_t><<<g512, 256, 0, stream>>>(Xv, Wvb, bv, Vb, 512, 512);

  // 3. flash attention
  attn_wmma<<<dim3(SEQ_L / 64, N_HEADS, 4), 128, 0, stream>>>(Qb, Kb, Vb, Ao);

  // 4. depthwise conv branch
  conv_dw<<<(M_ROWS * 512 + CB - 1) / CB, CB, 0, stream>>>(queries, conv_w, conv_b, Co);

  // 5. output projections into concatenated buffer (ld = 1024)
  gemm_wmma<bf16_t><<<g512, 256, 0, stream>>>(Ao, Woab, bo_attn, Cat, 512, 1024);
  gemm_wmma<bf16_t><<<g512, 256, 0, stream>>>(Co, Wocb, bo_conv, Cat + 512, 512, 1024);

  // 6. final projection (8192x1024x1024), f32 out
  gemm_wmma<float><<<dim3(1024 / 128, M_ROWS / 64), 256, 0, stream>>>(Cat, Wfb, bfin, Y, 1024, 1024);

  // 7. residual + layernorm
  add_ln<<<M_ROWS, 256, 0, stream>>>(queries, Y, gamma, beta, (float*)d_out);
}
